// ResidualAttnBlock_4157528342915
// MI455X (gfx1250) — matrix operand order repaired and checked by reading, not compiled, not run on hardware
//
#include <hip/hip_runtime.h>
#include <hip/hip_bf16.h>

typedef __attribute__((ext_vector_type(16))) _Float16 v16h;
typedef __attribute__((ext_vector_type(8)))  float    v8f;
typedef __attribute__((ext_vector_type(4)))  unsigned int u32x4;
typedef __attribute__((ext_vector_type(8)))  int      i32x8;
typedef __attribute__((ext_vector_type(4)))  int      i32x4;
typedef _Float16 h16;

#define B_  4
#define C_  128
#define N_  4096
#define G_  32
#define CG_ 4
#define EPSF 1e-5f

// ---------------- WMMA fragment helpers (CDNA5 16x16x32 f16 layouts) ----------------
__device__ __forceinline__ v16h load_a_frag(const h16* p, int ldr, int lane) {
  int row = lane & 15, half = lane >> 4;
  v16h a;
#pragma unroll
  for (int v = 0; v < 8; ++v) {
    int k = (v < 4) ? (half * 8 + 2 * v) : (16 + half * 8 + 2 * (v - 4));
    a[2 * v]     = p[row * ldr + k];
    a[2 * v + 1] = p[row * ldr + k + 1];
  }
  return a;
}

__device__ __forceinline__ v16h load_b_frag(const h16* p, int ldr, int lane) {
  int col = lane & 15, half = lane >> 4;
  v16h b;
#pragma unroll
  for (int v = 0; v < 8; ++v) {
    int k = (v < 4) ? (half * 8 + 2 * v) : (16 + half * 8 + 2 * (v - 4));
    b[2 * v]     = p[col * ldr + k];
    b[2 * v + 1] = p[col * ldr + k + 1];
  }
  return b;
}

__device__ __forceinline__ v8f wmma16(v16h a, v16h b, v8f c) {
  return __builtin_amdgcn_wmma_f32_16x16x32_f16(false, a, false, b, (short)0, c,
                                                false, false);
}

// ---------------- TDM: 2-D tile load, global f16 -> LDS (ISA §8.3/8.4 D# layout) ------
// rows x cols tile of 2-byte elements, row stride `row_stride` elements, rows tight in LDS.
__device__ __forceinline__ void tdm_load_2d(const h16* gsrc, h16* lds_dst,
                                            int rows, int cols, int row_stride) {
  unsigned long long ga = (unsigned long long)(size_t)gsrc;
  unsigned lds = (unsigned)(size_t)lds_dst;    // generic LDS addr low 32 bits == LDS offset
  u32x4 g0;
  g0[0] = 1u;                                                 // count=1, user D#
  g0[1] = lds;                                                // lds_addr
  g0[2] = (unsigned)ga;                                       // global_addr[31:0]
  g0[3] = (unsigned)((ga >> 32) & 0x01FFFFFFu) | (2u << 30);  // global_addr[56:32] | type=2
  i32x8 g1;
  g1[0] = 0x00010000;                            // wg_mask=0, data_size=1 (2 bytes)
  g1[1] = (int)((unsigned)cols << 16);           // atomic_barrier=0 | tensor_dim0[15:0]
  g1[2] = (int)((unsigned)N_ << 16);             // tensor_dim0[31:16]=0 | tensor_dim1[15:0]
  g1[3] = (int)((unsigned)cols << 16);           // tensor_dim1[31:16]=0 | tile_dim0
  g1[4] = rows;                                  // tile_dim1 | tile_dim2=0
  g1[5] = row_stride;                            // tensor_dim0_stride[31:0]
  g1[6] = 0;                                     // stride msbs | dim1_stride lsbs
  g1[7] = 0;
  i32x4 z4 = {0, 0, 0, 0};
  i32x8 z8 = {0, 0, 0, 0, 0, 0, 0, 0};
  // amdgpu-toolchain (clang-23) 6-arg form: (g0, g1, g2, g3, g4, cpol)
  __builtin_amdgcn_tensor_load_to_lds(g0, g1, z4, z4, z8, 0);
}

// ---------------- Kernel 1: GroupNorm, output transposed [B][N][C] f16 ----------------
__global__ void gn_kernel(const float* __restrict__ x, const float* __restrict__ w,
                          const float* __restrict__ bb, h16* __restrict__ h_t) {
  __shared__ float red[256];
  __shared__ float red2[256];
  int blk = blockIdx.x;
  int batch = blk / G_, g = blk % G_;
  const float* xg = x + ((size_t)batch * C_ + g * CG_) * N_;
  int tid = threadIdx.x;
  float s = 0.f, ss = 0.f;
  for (int i = tid; i < CG_ * N_; i += 256) {
    float v = xg[i];
    s += v; ss += v * v;
  }
  red[tid] = s; red2[tid] = ss;
  __syncthreads();
  for (int off = 128; off > 0; off >>= 1) {
    if (tid < off) { red[tid] += red[tid + off]; red2[tid] += red2[tid + off]; }
    __syncthreads();
  }
  float mean = red[0] * (1.f / (CG_ * N_));
  float var  = red2[0] * (1.f / (CG_ * N_)) - mean * mean;
  float inv  = rsqrtf(var + EPSF);
  for (int i = tid; i < CG_ * N_; i += 256) {
    int cc = i >> 12;
    int n  = i & (N_ - 1);
    int ch = g * CG_ + cc;
    float val = (xg[i] - mean) * inv * w[ch] + bb[ch];
    h_t[((size_t)batch * N_ + n) * C_ + ch] = (h16)val;
  }
}

// ---------------- Kernel 2: QKV GEMM (384x128 @ 128xN), WMMA, split q/k/v ----------------
__global__ void qkv_kernel(const h16* __restrict__ h_t, const float* __restrict__ qkv_w,
                           const float* __restrict__ qkv_b, h16* __restrict__ q_t,
                           h16* __restrict__ k_t, h16* __restrict__ v_t) {
  __shared__ h16 As[32 * 32];
  __shared__ h16 Bs[64 * 32];
  int n0 = blockIdx.x * 64;
  int d0 = blockIdx.y * 32;
  int batch = blockIdx.z;
  int tid = threadIdx.x;
  int wave = tid >> 5, lane = tid & 31;
  int wm = wave >> 2, wn = wave & 3;
  const h16* hb = h_t + (size_t)batch * N_ * C_;
  v8f acc = {};
  for (int c0 = 0; c0 < C_; c0 += 32) {
#pragma unroll
    for (int r = 0; r < 4; ++r) {
      int idx = tid * 4 + r;
      int row = idx >> 5, col = idx & 31;
      As[idx] = (h16)qkv_w[(d0 + row) * C_ + c0 + col];
    }
#pragma unroll
    for (int r = 0; r < 8; ++r) {
      int idx = tid * 8 + r;
      int row = idx >> 5, col = idx & 31;
      Bs[idx] = hb[(size_t)(n0 + row) * C_ + c0 + col];
    }
    __syncthreads();
    v16h a = load_a_frag(As + wm * 16 * 32, 32, lane);
    v16h b = load_b_frag(Bs + wn * 16 * 32, 32, lane);
    acc = wmma16(a, b, acc);
    __syncthreads();
  }
  int half = lane >> 4;
  int n = n0 + wn * 16 + (lane & 15);
  const float scale = 0.08838834764831845f;
  size_t off = ((size_t)batch * N_ + n) * C_;
#pragma unroll
  for (int j = 0; j < 8; ++j) {
    int d = d0 + wm * 16 + half * 8 + j;
    float val = acc[j] + qkv_b[d];
    if (d < C_)           q_t[off + d]          = (h16)(val * scale);
    else if (d < 2 * C_)  k_t[off + d - C_]     = (h16)val;
    else                  v_t[off + d - 2 * C_] = (h16)val;
  }
}

// ---------------- Kernel 3: flash attention, TDM-fed tiles ----------------
__global__ void attn_kernel(const h16* __restrict__ q_t, const h16* __restrict__ k_t,
                            const h16* __restrict__ v_t, h16* __restrict__ o_t) {
  __shared__ h16 Qs[64 * 128];      // [n_local][c]
  __shared__ h16 Ks[32 * 128];      // [m_local][c]
  __shared__ h16 Vt[128 * 32];      // [c][m_local]
  __shared__ h16 Ps[8][16 * 32];    // per-wave P staging
  int n0 = blockIdx.x * 64;
  int batch = blockIdx.y;
  int tid = threadIdx.x;
  int wave = tid >> 5, lane = tid & 31;
  int rg = wave >> 1, chh = wave & 1;
  const size_t base = (size_t)batch * N_ * C_;

  // Q tile via Tensor Data Mover (wave 0 issues; TENSORcnt drains; barrier publishes)
  if (wave == 0) {
    tdm_load_2d(q_t + base + (size_t)n0 * C_, Qs, 64, C_, C_);
    __builtin_amdgcn_s_wait_tensorcnt(0);
  }
  __syncthreads();

  float Mrun[8], Lrun[8];
#pragma unroll
  for (int j = 0; j < 8; ++j) { Mrun[j] = -1e30f; Lrun[j] = 0.f; }
  v8f O0 = {}, O1 = {}, O2 = {}, O3 = {};

  for (int m0 = 0; m0 < N_; m0 += 32) {
    // K tile: TDM async copy overlapped with the manual V transpose below
    if (wave == 0)
      tdm_load_2d(k_t + base + (size_t)m0 * C_, Ks, 32, C_, C_);
    {
      int m = tid >> 3, cb = (tid & 7) * 16;
#pragma unroll
      for (int c = 0; c < 16; ++c)
        Vt[(cb + c) * 32 + m] = v_t[base + (size_t)(m0 + m) * C_ + cb + c];
    }
    if (m0 + 32 < N_)   // prefetch next V tile into L2 (global_prefetch_b8)
      __builtin_prefetch(v_t + base + (size_t)(m0 + 32) * C_ + tid * 16, 0, 0);
    if (wave == 0)
      __builtin_amdgcn_s_wait_tensorcnt(0);
    __syncthreads();

    // S = Q * K^T
    v8f s0 = {}, s1 = {};
#pragma unroll
    for (int c0 = 0; c0 < C_; c0 += 32) {
      v16h a  = load_a_frag(Qs + rg * 16 * 128 + c0, 128, lane);
      v16h b0 = load_b_frag(Ks + c0, 128, lane);
      v16h b1 = load_b_frag(Ks + 16 * 128 + c0, 128, lane);
      s0 = wmma16(a, b0, s0);
      s1 = wmma16(a, b1, s1);
    }

    // online softmax
    float mnew[8];
#pragma unroll
    for (int j = 0; j < 8; ++j) mnew[j] = fmaxf(s0[j], s1[j]);
#pragma unroll
    for (int mask = 1; mask < 16; mask <<= 1)
#pragma unroll
      for (int j = 0; j < 8; ++j)
        mnew[j] = fmaxf(mnew[j], __shfl_xor(mnew[j], mask, 32));
    float alpha[8], rs[8];
#pragma unroll
    for (int j = 0; j < 8; ++j) {
      float Mn = fmaxf(Mrun[j], mnew[j]);
      alpha[j] = __expf(Mrun[j] - Mn);
      Mrun[j] = Mn;
      s0[j] = __expf(s0[j] - Mn);
      s1[j] = __expf(s1[j] - Mn);
      rs[j] = s0[j] + s1[j];
    }
#pragma unroll
    for (int mask = 1; mask < 16; mask <<= 1)
#pragma unroll
      for (int j = 0; j < 8; ++j) rs[j] += __shfl_xor(rs[j], mask, 32);
#pragma unroll
    for (int j = 0; j < 8; ++j) Lrun[j] = Lrun[j] * alpha[j] + rs[j];
#pragma unroll
    for (int j = 0; j < 8; ++j) {
      O0[j] *= alpha[j]; O1[j] *= alpha[j]; O2[j] *= alpha[j]; O3[j] *= alpha[j];
    }

    // stage P through per-wave LDS (D-layout -> A-layout)
    int half = lane >> 4, mcol = lane & 15;
    h16* ps = Ps[wave];
#pragma unroll
    for (int j = 0; j < 8; ++j) {
      ps[(half * 8 + j) * 32 + mcol]      = (h16)s0[j];
      ps[(half * 8 + j) * 32 + 16 + mcol] = (h16)s1[j];
    }
    __builtin_amdgcn_wave_barrier();
    asm volatile("s_wait_dscnt 0" ::: "memory");

    v16h ap  = load_a_frag(ps, 32, lane);
    v16h bv0 = load_b_frag(Vt + (chh * 64 + 0)  * 32, 32, lane);
    v16h bv1 = load_b_frag(Vt + (chh * 64 + 16) * 32, 32, lane);
    v16h bv2 = load_b_frag(Vt + (chh * 64 + 32) * 32, 32, lane);
    v16h bv3 = load_b_frag(Vt + (chh * 64 + 48) * 32, 32, lane);
    O0 = wmma16(ap, bv0, O0);
    O1 = wmma16(ap, bv1, O1);
    O2 = wmma16(ap, bv2, O2);
    O3 = wmma16(ap, bv3, O3);
    __syncthreads();
  }

  int half = lane >> 4, cl = lane & 15;
#pragma unroll
  for (int j = 0; j < 8; ++j) {
    float invl = 1.0f / Lrun[j];
    int row = n0 + rg * 16 + half * 8 + j;
    size_t off = base + (size_t)row * C_ + chh * 64 + cl;
    o_t[off + 0]  = (h16)(O0[j] * invl);
    o_t[off + 16] = (h16)(O1[j] * invl);
    o_t[off + 32] = (h16)(O2[j] * invl);
    o_t[off + 48] = (h16)(O3[j] * invl);
  }
}

// ---------------- Kernel 4: proj GEMM + bias + residual, f32 output ----------------
__global__ void proj_kernel(const h16* __restrict__ o_t, const float* __restrict__ proj_w,
                            const float* __restrict__ proj_b, const float* __restrict__ x,
                            float* __restrict__ out) {
  __shared__ h16 As[32 * 32];
  __shared__ h16 Bs[64 * 32];
  int n0 = blockIdx.x * 64;
  int d0 = blockIdx.y * 32;
  int batch = blockIdx.z;
  int tid = threadIdx.x, wave = tid >> 5, lane = tid & 31;
  int wm = wave >> 2, wn = wave & 3;
  const h16* ob = o_t + (size_t)batch * N_ * C_;
  v8f acc = {};
  for (int c0 = 0; c0 < C_; c0 += 32) {
#pragma unroll
    for (int r = 0; r < 4; ++r) {
      int idx = tid * 4 + r;
      int row = idx >> 5, col = idx & 31;
      As[idx] = (h16)proj_w[(d0 + row) * C_ + c0 + col];
    }
#pragma unroll
    for (int r = 0; r < 8; ++r) {
      int idx = tid * 8 + r;
      int row = idx >> 5, col = idx & 31;
      Bs[idx] = ob[(size_t)(n0 + row) * C_ + c0 + col];
    }
    __syncthreads();
    v16h a = load_a_frag(As + wm * 16 * 32, 32, lane);
    v16h b = load_b_frag(Bs + wn * 16 * 32, 32, lane);
    acc = wmma16(a, b, acc);
    __syncthreads();
  }
  int half = lane >> 4;
  int n = n0 + wn * 16 + (lane & 15);
#pragma unroll
  for (int j = 0; j < 8; ++j) {
    int d = d0 + wm * 16 + half * 8 + j;
    size_t off = ((size_t)batch * C_ + d) * N_ + n;
    out[off] = acc[j] + proj_b[d] + x[off];
  }
}

extern "C" void kernel_launch(void* const* d_in, const int* in_sizes, int n_in,
                              void* d_out, int out_size, void* d_ws, size_t ws_size,
                              hipStream_t stream) {
  const float* x      = (const float*)d_in[0];
  const float* norm_w = (const float*)d_in[1];
  const float* norm_b = (const float*)d_in[2];
  const float* qkv_w  = (const float*)d_in[3];
  const float* qkv_b  = (const float*)d_in[4];
  const float* proj_w = (const float*)d_in[5];
  const float* proj_b = (const float*)d_in[6];
  float* out = (float*)d_out;

  const size_t plane = (size_t)B_ * N_ * C_;
  h16* h_t = (h16*)d_ws;
  h16* q_t = h_t + plane;
  h16* k_t = q_t + plane;
  h16* v_t = k_t + plane;
  h16* o_t = v_t + plane;

  gn_kernel<<<dim3(B_ * G_), dim3(256), 0, stream>>>(x, norm_w, norm_b, h_t);
  qkv_kernel<<<dim3(N_ / 64, (3 * C_) / 32, B_), dim3(256), 0, stream>>>(
      h_t, qkv_w, qkv_b, q_t, k_t, v_t);
  attn_kernel<<<dim3(N_ / 64, B_), dim3(256), 0, stream>>>(q_t, k_t, v_t, o_t);
  proj_kernel<<<dim3(N_ / 64, C_ / 32, B_), dim3(256), 0, stream>>>(
      o_t, proj_w, proj_b, x, out);
}
